// GroupAttention_90941637526102
// MI455X (gfx1250) — compile-verified, matrix operand order checked by reading, not run on hardware
//
#include <hip/hip_runtime.h>
#include <hip/hip_bf16.h>
#include <stdint.h>

#define B_SZ   4
#define S_LEN  2048
#define D_MODEL 1024

typedef __attribute__((ext_vector_type(16))) __bf16 bf16x16;
typedef __attribute__((ext_vector_type(8)))  float  f32x8;
typedef __attribute__((ext_vector_type(4)))  unsigned int uint32x4;
typedef __attribute__((ext_vector_type(8)))  int  int32x8;
typedef __attribute__((ext_vector_type(4)))  int  int32x4;

__device__ __forceinline__ __bf16 f2bf(float f) {
  union { float f; uint32_t u; } in; in.f = f;
  uint32_t r = in.u + 0x7FFFu + ((in.u >> 16) & 1u);   // round-to-nearest-even
  union { uint16_t s; __bf16 b; } out; out.s = (uint16_t)(r >> 16);
  return out.b;
}
__device__ __forceinline__ float bf2f(uint16_t s) {
  union { uint32_t u; float f; } o; o.u = ((uint32_t)s) << 16; return o.f;
}

// ---------------- LayerNorm: context(f32) -> xn(bf16) ----------------
__global__ __launch_bounds__(256) void ln_kernel(const float* __restrict__ x,
    const float* __restrict__ gamma, const float* __restrict__ beta,
    __bf16* __restrict__ xn) {
  const int row = blockIdx.x;
  const int tid = threadIdx.x;
  const float4 v = reinterpret_cast<const float4*>(x + (size_t)row * D_MODEL)[tid];
  __shared__ float r0[256];
  __shared__ float r1[256];
  r0[tid] = v.x + v.y + v.z + v.w;
  r1[tid] = v.x*v.x + v.y*v.y + v.z*v.z + v.w*v.w;
  __syncthreads();
  for (int off = 128; off > 0; off >>= 1) {
    if (tid < off) { r0[tid] += r0[tid + off]; r1[tid] += r1[tid + off]; }
    __syncthreads();
  }
  const float inv = 1.0f / (float)D_MODEL;
  const float mu  = r0[0] * inv;
  const float var = r1[0] * inv - mu * mu;
  const float rs  = rsqrtf(var + 1e-5f);
  const float4 g  = reinterpret_cast<const float4*>(gamma)[tid];
  const float4 bt = reinterpret_cast<const float4*>(beta)[tid];
  __bf16* o = xn + (size_t)row * D_MODEL + tid * 4;
  o[0] = f2bf((v.x - mu) * rs * g.x + bt.x);
  o[1] = f2bf((v.y - mu) * rs * g.y + bt.y);
  o[2] = f2bf((v.z - mu) * rs * g.z + bt.z);
  o[3] = f2bf((v.w - mu) * rs * g.w + bt.w);
}

// ---------------- transpose+convert: Wq,Wk(f32 row-major) -> WqT,WkT(bf16) ----
__global__ __launch_bounds__(256) void wtrans_kernel(const float* __restrict__ Wq,
    const float* __restrict__ Wk, __bf16* __restrict__ WqT, __bf16* __restrict__ WkT) {
  const float* W  = blockIdx.z ? Wk  : Wq;
  __bf16*      WT = blockIdx.z ? WkT : WqT;
  __shared__ float tile[32][33];
  const int x0 = blockIdx.x * 32, y0 = blockIdx.y * 32;
  for (int j = threadIdx.y; j < 32; j += 8)
    tile[j][threadIdx.x] = W[(size_t)(y0 + j) * D_MODEL + x0 + threadIdx.x];
  __syncthreads();
  for (int j = threadIdx.y; j < 32; j += 8)
    WT[(size_t)(x0 + j) * D_MODEL + y0 + threadIdx.x] = f2bf(tile[threadIdx.x][j]);
}

// ---------------- TDM: load a rows x 32 bf16 tile (row stride K) into LDS ----
// D# packing per CDNA5 ISA:
//   group0: [1:0]=count=1, [63:32]=lds_addr, [120:64]=global_addr, [127:126]=type=2
//   group1: [17:16]=data_size(1 => 2B), [79:48]=tensor_dim0, [111:80]=tensor_dim1,
//           [127:112]=tile_dim0, [143:128]=tile_dim1, [207:160]=tensor_dim0_stride
__device__ __forceinline__ void tdm_load_b_tile(const __bf16* gsrc, uint32_t lds_off,
                                                int K, int Ndim) {
  const uint64_t ga = (uint64_t)(uintptr_t)gsrc;
  uint32x4 g0;
  g0[0] = 1u;                                            // count=1
  g0[1] = lds_off;                                       // lds byte address
  g0[2] = (uint32_t)(ga & 0xFFFFFFFFu);
  g0[3] = (uint32_t)((ga >> 32) & 0x01FFFFFFu) | (2u << 30);   // addr[56:32], type=2
  int32x8 g1;
  g1[0] = (int)(1u << 16);                               // data_size=1 (2 bytes)
  g1[1] = (int)(((uint32_t)K & 0xFFFFu) << 16);          // tensor_dim0[15:0]
  g1[2] = (int)(((uint32_t)K >> 16) | (((uint32_t)Ndim & 0xFFFFu) << 16));
  g1[3] = (int)(((uint32_t)Ndim >> 16) | (32u << 16));   // tile_dim0 = 32
  g1[4] = 64;                                            // tile_dim1 = 64, tile_dim2 = 0
  g1[5] = K;                                             // tensor_dim0_stride[31:0]
  g1[6] = 0;                                             // stride hi, dim1_stride lo
  g1[7] = 0;
  int32x4 gz = {0, 0, 0, 0};
#if defined(__clang_major__) && (__clang_major__ >= 23)
  int32x8 gz8 = {0, 0, 0, 0, 0, 0, 0, 0};
  __builtin_amdgcn_tensor_load_to_lds(g0, g1, gz, gz, gz8, 0);
#else
  __builtin_amdgcn_tensor_load_to_lds(g0, g1, gz, gz, 0);
#endif
}

// ---------------- WMMA GEMM: Out[m,n] = sum_k A[m,k] * Bsrc[n,k] --------------
// Per-lane operand layout per CDNA5 ISA 7.12.2 (16-bit A 16x32 / B 32x16):
// lane L (r=L&15, h=L>>4) holds two contiguous K-runs of 8 at offsets 8h, 16+8h.
// B tiles (shared by all 8 waves of the block) are staged into LDS by the TDM,
// double-buffered, synchronized with s_wait_tensorcnt + workgroup barriers.
__device__ __forceinline__ bf16x16 ldpair(const __bf16* p0, const __bf16* p1) {
  union { uint4 u[2]; bf16x16 v; } t;
  t.u[0] = *reinterpret_cast<const uint4*>(p0);
  t.u[1] = *reinterpret_cast<const uint4*>(p1);
  return t.v;
}

template <bool OUT_BF16>
__global__ __launch_bounds__(256) void gemm_bf16_wmma(
    const __bf16* __restrict__ A, const __bf16* __restrict__ Bsrc,
    void* __restrict__ Out, int Mdim, int Ndim, int K) {
  __shared__ __bf16 bstage[2][64 * 32];                  // 2 x 4KB double buffer
  const int lane = threadIdx.x & 31;
  const int wave = threadIdx.x >> 5;
  const int r = lane & 15;
  const int h = lane >> 4;
  const int m0 = blockIdx.y * 128 + wave * 16;
  const int n0 = blockIdx.x * 64;
  f32x8 acc0 = {}, acc1 = {}, acc2 = {}, acc3 = {};
  const __bf16* arow = A + (size_t)(m0 + r) * K;
  const uint32_t lds0 = (uint32_t)(uintptr_t)&bstage[0][0];  // LDS offset = addr[31:0]
  const uint32_t lds1 = (uint32_t)(uintptr_t)&bstage[1][0];

  if (wave == 0)                                          // prologue: fill buffer 0
    tdm_load_b_tile(Bsrc + (size_t)n0 * K, lds0, K, Ndim);

  for (int kk = 0; kk < K; kk += 32) {
    const int cur = (kk >> 5) & 1;
    if (wave == 0) {
      if (kk + 32 < K) {                                  // prefetch next buffer
        tdm_load_b_tile(Bsrc + (size_t)n0 * K + (kk + 32), cur ? lds0 : lds1, K, Ndim);
        __builtin_amdgcn_s_wait_tensorcnt(1);             // oldest (cur) complete
      } else {
        __builtin_amdgcn_s_wait_tensorcnt(0);
      }
    }
    __syncthreads();                                      // cur tile visible to all
    const __bf16* bs = &bstage[cur][0];
    const int o0 = kk + 8 * h;
    const int bo = r * 32 + 8 * h;                        // within 64x32 staged tile
    bf16x16 a  = ldpair(arow + o0, arow + o0 + 16);
    bf16x16 b0 = ldpair(bs + bo,             bs + bo + 16);
    bf16x16 b1 = ldpair(bs + bo + 16 * 32,   bs + bo + 16 * 32 + 16);
    bf16x16 b2 = ldpair(bs + bo + 32 * 32,   bs + bo + 32 * 32 + 16);
    bf16x16 b3 = ldpair(bs + bo + 48 * 32,   bs + bo + 48 * 32 + 16);
    acc0 = __builtin_amdgcn_wmma_f32_16x16x32_bf16(false, a, false, b0, (short)0, acc0, false, false);
    acc1 = __builtin_amdgcn_wmma_f32_16x16x32_bf16(false, a, false, b1, (short)0, acc1, false, false);
    acc2 = __builtin_amdgcn_wmma_f32_16x16x32_bf16(false, a, false, b2, (short)0, acc2, false, false);
    acc3 = __builtin_amdgcn_wmma_f32_16x16x32_bf16(false, a, false, b3, (short)0, acc3, false, false);
    __syncthreads();                                      // reads done before overwrite
  }
  // C/D f32 16x16 layout: VGPR j -> M = j + 8h, N = r
  #pragma unroll
  for (int j = 0; j < 8; ++j) {
    const int m = m0 + j + 8 * h;
    const size_t rowoff = (size_t)m * Ndim + n0 + r;
    if constexpr (OUT_BF16) {
      __bf16* o = reinterpret_cast<__bf16*>(Out);
      o[rowoff]      = f2bf(acc0[j]);
      o[rowoff + 16] = f2bf(acc1[j]);
      o[rowoff + 32] = f2bf(acc2[j]);
      o[rowoff + 48] = f2bf(acc3[j]);
    } else {
      float* o = reinterpret_cast<float*>(Out);
      o[rowoff]      = acc0[j];
      o[rowoff + 16] = acc1[j];
      o[rowoff + 32] = acc2[j];
      o[rowoff + 48] = acc3[j];
    }
  }
}

// ---------------- band scores + 2-entry softmax ------------------------------
// sc_up[s] = xn[s].z[s+1]/32 ; sc_dn[s] = xn[s].z[s-1]/32 ; masked -> -1e9
__global__ __launch_bounds__(256) void band_scores_kernel(
    const __bf16* __restrict__ xn, const float* __restrict__ z,
    const long long* __restrict__ eos, float* __restrict__ sm_up,
    float* __restrict__ sm_dn) {
  const int row = blockIdx.x;          // b*S + s
  const int s = row & (S_LEN - 1);
  const int tid = threadIdx.x;
  const ushort4 xv = reinterpret_cast<const ushort4*>(xn + (size_t)row * D_MODEL)[tid];
  const float x0 = bf2f(xv.x), x1 = bf2f(xv.y), x2 = bf2f(xv.z), x3 = bf2f(xv.w);
  float du = 0.f, dd = 0.f;
  if (s + 1 < S_LEN) {
    const float4 zv = reinterpret_cast<const float4*>(z + (size_t)(row + 1) * D_MODEL)[tid];
    du = x0 * zv.x + x1 * zv.y + x2 * zv.z + x3 * zv.w;
  }
  if (s > 0) {
    const float4 zv = reinterpret_cast<const float4*>(z + (size_t)(row - 1) * D_MODEL)[tid];
    dd = x0 * zv.x + x1 * zv.y + x2 * zv.z + x3 * zv.w;
  }
  __shared__ float su[256];
  __shared__ float sd[256];
  su[tid] = du; sd[tid] = dd;
  __syncthreads();
  for (int off = 128; off > 0; off >>= 1) {
    if (tid < off) { su[tid] += su[tid + off]; sd[tid] += sd[tid + off]; }
    __syncthreads();
  }
  if (tid == 0) {
    const float scale = 0.03125f;      // 1/sqrt(1024)
    float scu = -1e9f, scd = -1e9f;
    if (s + 1 < S_LEN && eos[(size_t)row * S_LEN + (s + 1)] != 0) scu = su[0] * scale;
    if (s > 0        && eos[(size_t)row * S_LEN + (s - 1)] != 0) scd = sd[0] * scale;
    const float m  = fmaxf(scu, scd);
    const float eu = expf(scu - m), ed = expf(scd - m);
    const float Zi = 1.0f / (eu + ed);
    sm_up[row] = eu * Zi;
    sm_dn[row] = ed * Zi;
  }
}

// ---------------- band neibor values + per-batch prefix sum P ----------------
__global__ __launch_bounds__(256) void scan_kernel(
    const float* __restrict__ sm_up, const float* __restrict__ sm_dn,
    const float* __restrict__ prior, float* __restrict__ v_up,
    float* __restrict__ P) {
  const int b = blockIdx.x;
  const int tid = threadIdx.x;
  const int CH = S_LEN / 256;          // 8
  const int base = b * S_LEN + tid * CH;
  float Lloc[8];
  float acc = 0.f;
  #pragma unroll
  for (int j = 0; j < CH; ++j) {
    const int s = tid * CH + j;
    float Lv = 0.f;
    if (s < S_LEN - 1) {
      const float v = sqrtf(sm_up[base + j] * sm_dn[base + j + 1] + 1e-9f);
      v_up[base + j] = v;
      const float p = prior[(size_t)b * S_LEN * S_LEN + (size_t)s * S_LEN + (s + 1)];
      Lv = logf(p + (1.f - p) * v + 1e-9f);
    } else {
      v_up[base + j] = 0.f;
    }
    Lloc[j] = Lv;
    acc += Lv;
  }
  __shared__ float sbuf[256];
  sbuf[tid] = acc;
  __syncthreads();
  for (int off = 1; off < 256; off <<= 1) {
    const float y = (tid >= off) ? sbuf[tid - off] : 0.f;
    __syncthreads();
    sbuf[tid] += y;
    __syncthreads();
  }
  float run = sbuf[tid] - acc;         // exclusive prefix of per-thread sums
  #pragma unroll
  for (int j = 0; j < CH; ++j) {
    run += Lloc[j];
    P[base + j] = run;                 // inclusive cumsum of L
  }
}

// ---------------- dense fill of both outputs (HBM-bound) ---------------------
// g_attn[s>t] = exp(P[s]-P[t-2]) + 1e-9 ; symmetric for s<t ; diag = neibor diag
// neibor[s,t] = prior + (1-prior)*( band ? v_up : sqrt(1e-9) )
__global__ __launch_bounds__(256) void fill_kernel(
    const float* __restrict__ prior, const float* __restrict__ v_up,
    const float* __restrict__ P, float* __restrict__ g_out,
    float* __restrict__ n_out) {
  const float CSQ = 3.1622776601683795e-5f;  // sqrt(1e-9)
  const int gid = blockIdx.x * 256 + threadIdx.x;
  const int TQ = S_LEN / 4;
  const int srow = gid / TQ;                 // b*S + s
  if (srow >= B_SZ * S_LEN) return;
  const int tq = gid - srow * TQ;
  const int s = srow & (S_LEN - 1);
  const int b = srow >> 11;
  const size_t base = (size_t)srow * S_LEN + (size_t)tq * 4;
  __builtin_prefetch(prior + base + 16384, 0, 0);
  const float4 pr = *reinterpret_cast<const float4*>(prior + base);
  const float prc[4] = { pr.x, pr.y, pr.z, pr.w };
  const float* Pb = P + b * S_LEN;
  const float* Vb = v_up + b * S_LEN;
  const float Ps   = Pb[s];
  const float Psm2 = (s >= 2) ? Pb[s - 2] : 0.f;
  float gv[4], nv[4];
  #pragma unroll
  for (int c = 0; c < 4; ++c) {
    const int t = tq * 4 + c;
    const float p = prc[c];
    float sym = CSQ;
    if (t == s + 1)      sym = Vb[s];
    else if (t + 1 == s) sym = Vb[t];
    nv[c] = p + (1.f - p) * sym;
    if (s > t)      gv[c] = expf(Ps - ((t >= 2) ? Pb[t - 2] : 0.f)) + 1e-9f;
    else if (s < t) gv[c] = expf(Pb[t] - Psm2) + 1e-9f;
    else            gv[c] = nv[c];
  }
  *reinterpret_cast<float4*>(g_out + base) = make_float4(gv[0], gv[1], gv[2], gv[3]);
  *reinterpret_cast<float4*>(n_out + base) = make_float4(nv[0], nv[1], nv[2], nv[3]);
}

extern "C" void kernel_launch(void* const* d_in, const int* in_sizes, int n_in,
                              void* d_out, int out_size, void* d_ws, size_t ws_size,
                              hipStream_t stream) {
  const float*     context = (const float*)d_in[0];
  const long long* eos     = (const long long*)d_in[1];   // int64
  const float*     prior   = (const float*)d_in[2];
  const float*     Wq      = (const float*)d_in[3];
  const float*     Wk      = (const float*)d_in[4];
  const float*     gamma   = (const float*)d_in[5];
  const float*     beta    = (const float*)d_in[6];

  const size_t M = (size_t)B_SZ * S_LEN;                  // 8192 rows
  char* ws = (char*)d_ws;
  size_t off = 0;
  auto carve = [&](size_t bytes) {
    char* p = ws + off;
    off = (off + bytes + 255) & ~(size_t)255;
    return (void*)p;
  };
  __bf16* xn    = (__bf16*)carve(M * D_MODEL * 2);                      // 16 MB
  __bf16* WqT   = (__bf16*)carve((size_t)D_MODEL * D_MODEL * 2);        //  2 MB
  __bf16* WkT   = (__bf16*)carve((size_t)D_MODEL * D_MODEL * 2);        //  2 MB
  __bf16* Mmat  = (__bf16*)carve((size_t)D_MODEL * D_MODEL * 2);        //  2 MB
  float*  z     = (float*) carve(M * D_MODEL * 4);                      // 32 MB
  float*  sm_up = (float*) carve(M * 4);
  float*  sm_dn = (float*) carve(M * 4);
  float*  v_up  = (float*) carve(M * 4);
  float*  P     = (float*) carve(M * 4);
  (void)ws_size; (void)in_sizes; (void)n_in; (void)out_size;

  float* g_out = (float*)d_out;
  float* n_out = g_out + M * S_LEN;

  // 1) bf16 transposes of Wq, Wk
  wtrans_kernel<<<dim3(D_MODEL / 32, D_MODEL / 32, 2), dim3(32, 8, 1), 0, stream>>>(
      Wq, Wk, WqT, WkT);
  // 2) layernorm -> xn bf16
  ln_kernel<<<dim3((unsigned)M), dim3(256), 0, stream>>>(context, gamma, beta, xn);
  // 3) Mmat[i,j] = sum_n Wq[n,i]*Wk[n,j]   (1024^3, WMMA + TDM-staged B)
  gemm_bf16_wmma<true><<<dim3(D_MODEL / 64, D_MODEL / 128), dim3(256), 0, stream>>>(
      WqT, WkT, (void*)Mmat, D_MODEL, D_MODEL, D_MODEL);
  // 4) z[s,i] = sum_j xn[s,j]*Mmat[i,j]    (8192x1024x1024, WMMA + TDM-staged B)
  gemm_bf16_wmma<false><<<dim3(D_MODEL / 64, (unsigned)(M / 128)), dim3(256), 0, stream>>>(
      xn, Mmat, (void*)z, (int)M, D_MODEL, D_MODEL);
  // 5) band scores + softmax
  band_scores_kernel<<<dim3((unsigned)M), dim3(256), 0, stream>>>(xn, z, eos, sm_up, sm_dn);
  // 6) band neibor + prefix sums
  scan_kernel<<<dim3(B_SZ), dim3(256), 0, stream>>>(sm_up, sm_dn, prior, v_up, P);
  // 7) dense output fill (HBM-bound, ~210 MB)
  fill_kernel<<<dim3((unsigned)((M * S_LEN / 4 + 255) / 256)), dim3(256), 0, stream>>>(
      prior, v_up, P, g_out, n_out);
}
